// AttnNN_75548474737070
// MI455X (gfx1250) — matrix operand order repaired and checked by reading, not compiled, not run on hardware
//
#include <hip/hip_runtime.h>
#include <hip/hip_bf16.h>

// ---------- types ----------
typedef __attribute__((ext_vector_type(16))) __bf16        v16bf;
typedef __attribute__((ext_vector_type(8)))  float         v8f;
typedef __attribute__((ext_vector_type(4)))  unsigned int  u32x4;
typedef __attribute__((ext_vector_type(2)))  unsigned int  u32x2;
typedef __attribute__((ext_vector_type(4)))  float         f32x4;

union Frag16 { u32x4 u[2]; v16bf v; };   // 32 bytes = 16 bf16 per lane

// round-to-nearest-even fp32 -> bf16
__device__ __forceinline__ unsigned int bf1(float a) {
  unsigned int ua = __float_as_uint(a);
  return (ua + 0x7FFFu + ((ua >> 16) & 1u)) >> 16;
}
__device__ __forceinline__ unsigned int pack2bf(float a, float b) {
  return bf1(a) | (bf1(b) << 16);
}
__device__ __forceinline__ float bf2f(unsigned short h) {
  return __uint_as_float((unsigned int)h << 16);
}

// =====================================================================
// Fused WMMA GEMM:  E[M,N] = A[M,K] * B[N,K]^T   (bf16 A, f32 acc)
//   ACT==0 : write raw E to out[M,N]                     (GRU gates)
//   ACT==1 : scores[r] = v_b + sum_n v[n]*softplus(E+bias[b,n])
//   ACT==2 : same with tanh                              (final attn)
// A (bf16 in global) is staged into LDS in 512-col K-segments via
// ASYNC global->LDS DMA, software-pipelined against the WMMA loop
// (s_wait_asynccnt; loads complete in order: 16 per wave per segment).
// B fragments stream straight from global (bf16 if BF16B, else fp32+cvt).
// 256 threads = 8 waves; wave -> one 16-wide n tile, 4 m tiles of 16.
// =====================================================================
template<int ACT, int BF16B>
__global__ __launch_bounds__(256) void wmma_gemm(
    const unsigned short* __restrict__ Abf, int lda, int K,
    const void* __restrict__ Bw, int N,
    const float* __restrict__ bias,       // [64,N] or null
    const float* __restrict__ vw,         // [N]     (ACT>0)
    const float* __restrict__ vb,         // [1]     (ACT>0)
    float* __restrict__ out,
    int nChunksPerWG)
{
  __shared__ __align__(16) unsigned short As[64 * 2048];  // 256 KB (<=320KB WGP LDS)
  __shared__ float sAcc[64];

  const int tid  = threadIdx.x;
  const int wave = tid >> 5, lane = tid & 31;
  const int rowBase = blockIdx.x * 64;
  const int nseg = K >> 9;                 // 512-wide K segments

  // async-issue one K segment: 64 rows x 512 bf16 = 64KB = 16 b128/lane/wave
  auto issue_seg = [&](int s) {
    #pragma unroll
    for (int it = 0; it < 16; ++it) {
      int unitIdx = it * 32 + lane;                 // 0..511 (b128 units)
      int r = wave * 8 + (unitIdx >> 6);            // 8 rows per wave
      int u = unitIdx & 63;                         // 64 b128 units per row-seg
      // ISA 10.2: LDS aperture -> LDS offset is the low 32 bits of the address
      unsigned ldsAddr = (unsigned)(size_t)(&As[r * 2048 + s * 512 + u * 8]);
      unsigned long long ga =
          (unsigned long long)(size_t)(Abf + (size_t)(rowBase + r) * lda + s * 512 + u * 8);
      asm volatile("global_load_async_to_lds_b128 %0, %1, off"
                   :: "v"(ldsAddr), "v"(ga) : "memory");
    }
  };

  if (ACT != 0 && tid < 64) sAcc[tid] = 0.f;
  issue_seg(0);

  const int lhalf = lane >> 4, lmod = lane & 15;

  for (int c = 0; c < nChunksPerWG; ++c) {
    const int nbase = ((blockIdx.y * nChunksPerWG + c) * 8 + wave) * 16;
    v8f acc[4] = {};
    const size_t browOff = (size_t)(nbase + lmod) * K;

    for (int s = 0; s < nseg; ++s) {
      if (c == 0) {                         // pipelined staging on first n-chunk
        if (s + 1 < nseg) {
          issue_seg(s + 1);                 // prefetch next segment
          asm volatile("s_wait_asynccnt 0x10" ::: "memory"); // seg s done (in-order)
        } else {
          asm volatile("s_wait_asynccnt 0x0" ::: "memory");
        }
        __syncthreads();                    // all waves' portions visible
      }

      for (int kb = s * 512; kb < s * 512 + 512; kb += 32) {
        // B fragment: column n = nbase+lmod; half-wave K split, K runs [16*lhalf,+16) as for A
        Frag16 bfr;
        if (BF16B) {
          const unsigned short* p = (const unsigned short*)Bw + browOff + kb + lhalf * 16;
          bfr.u[0] = *(const u32x4*)p;
          bfr.u[1] = *(const u32x4*)(p + 8);
        } else {
          const float* p = (const float*)Bw + browOff + kb + lhalf * 16;
          f32x4 a0 = *(const f32x4*)p,       a1 = *(const f32x4*)(p + 4);
          f32x4 b0 = *(const f32x4*)(p + 8), b1 = *(const f32x4*)(p + 12);
          u32x4 t0; t0.x = pack2bf(a0.x, a0.y); t0.y = pack2bf(a0.z, a0.w);
                    t0.z = pack2bf(a1.x, a1.y); t0.w = pack2bf(a1.z, a1.w);
          u32x4 t1; t1.x = pack2bf(b0.x, b0.y); t1.y = pack2bf(b0.z, b0.w);
                    t1.z = pack2bf(b1.x, b1.y); t1.w = pack2bf(b1.z, b1.w);
          bfr.u[0] = t0; bfr.u[1] = t1;
        }
        #pragma unroll
        for (int mt = 0; mt < 4; ++mt) {
          // A fragment: row m = mt*16 + lmod; K runs [16*lhalf,+16)
          Frag16 afr;
          const unsigned short* ap = &As[(mt * 16 + lmod) * 2048 + kb + lhalf * 16];
          afr.u[0] = *(const u32x4*)ap;
          afr.u[1] = *(const u32x4*)(ap + 8);
          acc[mt] = __builtin_amdgcn_wmma_f32_16x16x32_bf16(
              false, afr.v, false, bfr.v, (short)0, acc[mt], false, false);
        }
      }
    }

    if (ACT == 0) {
      #pragma unroll
      for (int mt = 0; mt < 4; ++mt)
        #pragma unroll
        for (int i = 0; i < 8; ++i) {
          int m = rowBase + mt * 16 + i + lhalf * 8;   // C/D layout: elem i <-> M=i+8*half
          out[(size_t)m * N + nbase + lmod] = acc[mt][i];
        }
    } else {
      float vn = vw[nbase + lmod];
      #pragma unroll
      for (int mt = 0; mt < 4; ++mt) {
        #pragma unroll
        for (int i = 0; i < 8; ++i) {
          int ml = mt * 16 + i + lhalf * 8;            // local row 0..63
          int b  = (rowBase + ml) & 63;                // rows are s*64+b
          float x = acc[mt][i] + bias[b * N + nbase + lmod];
          float e = (ACT == 1) ? ((x > 20.f) ? x : __logf(1.f + __expf(x)))
                               : tanhf(x);
          float contrib = e * vn;
          contrib += __shfl_xor(contrib, 1);
          contrib += __shfl_xor(contrib, 2);
          contrib += __shfl_xor(contrib, 4);
          contrib += __shfl_xor(contrib, 8);           // reduce over n in 16-lane group
          if (lmod == 0) atomicAdd(&sAcc[ml], contrib); // ds_add_f32
        }
      }
    }
  }

  if (ACT != 0) {
    __syncthreads();
    if (tid < 64) out[rowBase + tid] = sAcc[tid] + vb[0];
  }
}

// ---------- fp32 -> bf16 conversion ----------
__global__ __launch_bounds__(256) void cvt_bf16_kernel(
    const float* __restrict__ in, unsigned short* __restrict__ out, int n) {
  int i = (blockIdx.x * 256 + threadIdx.x) * 4;
  if (i >= n) return;
  f32x4 f = *(const f32x4*)(in + i);
  u32x2 p; p.x = pack2bf(f.x, f.y); p.y = pack2bf(f.z, f.w);
  *(u32x2*)(out + i) = p;
}

// ---------- bias[b,h] = Wh_b[h] + Ws_b[h] + d_hidden[b,:]·Ws_w[h,:] ----------
__global__ __launch_bounds__(256) void bias_kernel(
    const float* __restrict__ dh, const float* __restrict__ Wsw,
    const float* __restrict__ Wsb, const float* __restrict__ Whb,
    float* __restrict__ out) {
  int gid = blockIdx.x * 256 + threadIdx.x;   // 64*1024
  int b = gid >> 10, h = gid & 1023;
  const float* xr = dh + b * 1024;
  const float* wr = Wsw + (size_t)h * 1024;
  float s = 0.f;
  for (int k = 0; k < 1024; k += 4) {
    f32x4 x = *(const f32x4*)(xr + k);
    f32x4 w = *(const f32x4*)(wr + k);
    s += x.x * w.x + x.y * w.y + x.z * w.z + x.w * w.w;
  }
  out[gid] = s + Wsb[h] + Whb[h];
}

// ---------- softmax over s: scores[S,64] -> attn[64,S] ----------
__global__ __launch_bounds__(256) void softmax_kernel(
    const float* __restrict__ scores, float* __restrict__ attn, int S) {
  int b = blockIdx.x, tid = threadIdx.x;
  __shared__ float red[256];
  float mx = -3.4e38f;
  for (int s = tid; s < S; s += 256) mx = fmaxf(mx, scores[s * 64 + b]);
  red[tid] = mx; __syncthreads();
  for (int off = 128; off > 0; off >>= 1) {
    if (tid < off) red[tid] = fmaxf(red[tid], red[tid + off]);
    __syncthreads();
  }
  mx = red[0]; __syncthreads();
  float sum = 0.f;
  for (int s = tid; s < S; s += 256) {
    float e = __expf(scores[s * 64 + b] - mx);
    attn[b * S + s] = e; sum += e;
  }
  red[tid] = sum; __syncthreads();
  for (int off = 128; off > 0; off >>= 1) {
    if (tid < off) red[tid] += red[tid + off];
    __syncthreads();
  }
  float inv = 1.f / red[0];
  for (int s = tid; s < S; s += 256) attn[b * S + s] *= inv;
}

// ---------- ctx[b,d] = sum_s attn[b,s] * outs_bf[s,b,d]  (bf16 in) ----------
template<int OUTBF>
__global__ __launch_bounds__(256) void context_kernel(
    const float* __restrict__ attn, const unsigned short* __restrict__ outs,
    void* __restrict__ ctx, int S, int D) {
  int b = blockIdx.x;
  int d = blockIdx.y * 256 + threadIdx.x;
  float acc = 0.f;
  for (int s = 0; s < S; ++s)
    acc += attn[b * S + s] * bf2f(outs[((size_t)s * 64 + b) * D + d]);
  if (OUTBF)
    ((unsigned short*)ctx)[(size_t)b * D + d] = (unsigned short)bf1(acc);
  else
    ((float*)ctx)[(size_t)b * D + d] = acc;
}

// ---------- GRU pointwise combine; writes fp32 state + bf16 rep slot ----------
__global__ __launch_bounds__(256) void gru_pointwise(
    const float* __restrict__ gi, const float* __restrict__ gh,
    const float* __restrict__ bih, const float* __restrict__ bhh,
    const float* __restrict__ hprev, int hprev_stride,
    float* __restrict__ hnew_f,
    unsigned short* __restrict__ hnew_bf, int bf_stride) {
  int gid = blockIdx.x * 256 + threadIdx.x;   // 64*1024
  int b = gid >> 10, j = gid & 1023;
  float ir = gi[b * 3072 + j]        + bih[j];
  float iz = gi[b * 3072 + 1024 + j] + bih[1024 + j];
  float in_= gi[b * 3072 + 2048 + j] + bih[2048 + j];
  float hr = gh[b * 3072 + j]        + bhh[j];
  float hz = gh[b * 3072 + 1024 + j] + bhh[1024 + j];
  float hn = gh[b * 3072 + 2048 + j] + bhh[2048 + j];
  float r = 1.f / (1.f + __expf(-(ir + hr)));
  float z = 1.f / (1.f + __expf(-(iz + hz)));
  float n = tanhf(in_ + r * hn);
  float hp = hprev[b * hprev_stride + j];
  float hv = (1.f - z) * n + z * hp;
  hnew_f[b * 1024 + j] = hv;
  hnew_bf[b * bf_stride + j] = (unsigned short)bf1(hv);
}

// =====================================================================
extern "C" void kernel_launch(void* const* d_in, const int* in_sizes, int n_in,
                              void* d_out, int out_size, void* d_ws, size_t ws_size,
                              hipStream_t stream) {
  const float* d_hidden  = (const float*)d_in[0];
  const float* outputs_t = (const float*)d_in[1];
  const float* outputs_a = (const float*)d_in[2];
  const float* hidden_c  = (const float*)d_in[3];
  const float* Wh_t_w = (const float*)d_in[4];  const float* Wh_t_b = (const float*)d_in[5];
  const float* Ws_t_w = (const float*)d_in[6];  const float* Ws_t_b = (const float*)d_in[7];
  const float* v_t_w  = (const float*)d_in[8];  const float* v_t_b  = (const float*)d_in[9];
  const float* Wh_a_w = (const float*)d_in[10]; const float* Wh_a_b = (const float*)d_in[11];
  const float* Ws_a_w = (const float*)d_in[12]; const float* Ws_a_b = (const float*)d_in[13];
  const float* v_a_w  = (const float*)d_in[14]; const float* v_a_b  = (const float*)d_in[15];
  const float* wt_w   = (const float*)d_in[16]; const float* wt_b   = (const float*)d_in[17];
  const float* ws_w   = (const float*)d_in[18]; const float* ws_b   = (const float*)d_in[19];
  const float* v_w    = (const float*)d_in[20]; const float* v_b    = (const float*)d_in[21];
  const float* gwihf = (const float*)d_in[22]; const float* gwhhf = (const float*)d_in[23];
  const float* gbihf = (const float*)d_in[24]; const float* gbhhf = (const float*)d_in[25];
  const float* gwihr = (const float*)d_in[26]; const float* gwhhr = (const float*)d_in[27];
  const float* gbihr = (const float*)d_in[28]; const float* gbhhr = (const float*)d_in[29];

  char* ws = (char*)d_ws;
  size_t off = 0;
  auto take = [&](size_t bytes) -> char* {
    char* p = ws + off; off += (bytes + 255) & ~(size_t)255; return p;
  };
  unsigned short* out_t_bf = (unsigned short*)take((size_t)512 * 64 * 2048 * 2); // 128MB
  unsigned short* out_a_bf = (unsigned short*)take((size_t)64 * 64 * 2048 * 2);  // 16MB
  unsigned short* Wt_bf = (unsigned short*)take(1024 * 2048 * 2);
  unsigned short* Wa_bf = (unsigned short*)take(1024 * 2048 * 2);
  unsigned short* Wf_bf = (unsigned short*)take(1024 * 2048 * 2);
  unsigned short* hid_bf   = (unsigned short*)take(2 * 64 * 1024 * 2);
  unsigned short* ctx_t_bf = (unsigned short*)take(64 * 2048 * 2);
  unsigned short* ctx_a_bf = (unsigned short*)take(64 * 2048 * 2);
  unsigned short* rep_bf   = (unsigned short*)take(2 * 64 * 2048 * 2);  // [2][64][2048]
  float* bias_t   = (float*)take(64 * 1024 * 4);
  float* bias_a   = (float*)take(64 * 1024 * 4);
  float* bias_f   = (float*)take(64 * 1024 * 4);
  float* scores_t = (float*)take(512 * 64 * 4);
  float* scores_a = (float*)take(64 * 64 * 4);
  float* scores_f = (float*)take(2 * 64 * 4);
  float* gi       = (float*)take(64 * 3072 * 4);
  float* gh       = (float*)take(64 * 3072 * 4);
  float* h1f      = (float*)take(64 * 1024 * 4);
  float* h2f      = (float*)take(64 * 1024 * 4);
  float* r1f      = (float*)take(64 * 1024 * 4);
  float* r2f      = (float*)take(64 * 1024 * 4);

  float* out_attn_t = (float*)d_out;                   // [64,512]
  float* out_attn_a = out_attn_t + 64 * 512;           // [64,64]
  float* out_ctx    = out_attn_a + 64 * 64;            // [64,2048]
  float* out_attn   = out_ctx + 64 * 2048;             // [64,2]

  // --- bf16 conversions (GEMM A operands + weights) ---
  cvt_bf16_kernel<<<32768, 256, 0, stream>>>(outputs_t, out_t_bf, 512 * 64 * 2048);
  cvt_bf16_kernel<<<4096, 256, 0, stream>>>(outputs_a, out_a_bf, 64 * 64 * 2048);
  cvt_bf16_kernel<<<2048, 256, 0, stream>>>(Wh_t_w, Wt_bf, 1024 * 2048);
  cvt_bf16_kernel<<<2048, 256, 0, stream>>>(Wh_a_w, Wa_bf, 1024 * 2048);
  cvt_bf16_kernel<<<2048, 256, 0, stream>>>(wt_w,   Wf_bf, 1024 * 2048);
  cvt_bf16_kernel<<<128, 256, 0, stream>>>(hidden_c, hid_bf, 2 * 64 * 1024);

  // --- per-(b,h) combined biases: Wh_b + Ws_b + d_hidden·Ws_w^T ---
  bias_kernel<<<256, 256, 0, stream>>>(d_hidden, Ws_t_w, Ws_t_b, Wh_t_b, bias_t);
  bias_kernel<<<256, 256, 0, stream>>>(d_hidden, Ws_a_w, Ws_a_b, Wh_a_b, bias_a);
  bias_kernel<<<256, 256, 0, stream>>>(d_hidden, ws_w,   ws_b,   wt_b,   bias_f);

  // --- big fused attention-score GEMMs (softplus + v-dot epilogue) ---
  wmma_gemm<1, 1><<<dim3(512, 1), 256, 0, stream>>>(
      out_t_bf, 2048, 2048, Wt_bf, 1024, bias_t, v_t_w, v_t_b, scores_t, 8);
  wmma_gemm<1, 1><<<dim3(64, 1), 256, 0, stream>>>(
      out_a_bf, 2048, 2048, Wa_bf, 1024, bias_a, v_a_w, v_a_b, scores_a, 8);

  softmax_kernel<<<64, 256, 0, stream>>>(scores_t, out_attn_t, 512);
  softmax_kernel<<<64, 256, 0, stream>>>(scores_a, out_attn_a, 64);
  context_kernel<1><<<dim3(64, 8), 256, 0, stream>>>(out_attn_t, out_t_bf, ctx_t_bf, 512, 2048);
  context_kernel<1><<<dim3(64, 8), 256, 0, stream>>>(out_attn_a, out_a_bf, ctx_a_bf, 64, 2048);

  // --- bidirectional GRU over 2 steps; bf16 states land in rep slots ---
  unsigned short* h1bf = rep_bf;                    // rep[0][:,   0:1024]
  unsigned short* h2bf = rep_bf + 131072;           // rep[1][:,   0:1024]
  unsigned short* r1bf = rep_bf + 131072 + 1024;    // rep[1][:,1024:2048]
  unsigned short* r2bf = rep_bf + 1024;             // rep[0][:,1024:2048]

  // h1 = cell(ctx_t, hidden_c[0], fwd)
  wmma_gemm<0, 0><<<dim3(1, 24), 256, 0, stream>>>(ctx_t_bf, 2048, 2048, gwihf, 3072, nullptr, nullptr, nullptr, gi, 1);
  wmma_gemm<0, 0><<<dim3(1, 24), 256, 0, stream>>>(hid_bf,   1024, 1024, gwhhf, 3072, nullptr, nullptr, nullptr, gh, 1);
  gru_pointwise<<<256, 256, 0, stream>>>(gi, gh, gbihf, gbhhf, hidden_c, 1024, h1f, h1bf, 2048);
  // h2 = cell(ctx_a, h1, fwd)
  wmma_gemm<0, 0><<<dim3(1, 24), 256, 0, stream>>>(ctx_a_bf, 2048, 2048, gwihf, 3072, nullptr, nullptr, nullptr, gi, 1);
  wmma_gemm<0, 0><<<dim3(1, 24), 256, 0, stream>>>(h1bf,     2048, 1024, gwhhf, 3072, nullptr, nullptr, nullptr, gh, 1);
  gru_pointwise<<<256, 256, 0, stream>>>(gi, gh, gbihf, gbhhf, h1f, 1024, h2f, h2bf, 2048);
  // r1 = cell(ctx_a, hidden_c[1], rev)
  wmma_gemm<0, 0><<<dim3(1, 24), 256, 0, stream>>>(ctx_a_bf, 2048, 2048, gwihr, 3072, nullptr, nullptr, nullptr, gi, 1);
  wmma_gemm<0, 0><<<dim3(1, 24), 256, 0, stream>>>(hid_bf + 64 * 1024, 1024, 1024, gwhhr, 3072, nullptr, nullptr, nullptr, gh, 1);
  gru_pointwise<<<256, 256, 0, stream>>>(gi, gh, gbihr, gbhhr, hidden_c + 64 * 1024, 1024, r1f, r1bf, 2048);
  // r2 = cell(ctx_t, r1, rev)
  wmma_gemm<0, 0><<<dim3(1, 24), 256, 0, stream>>>(ctx_t_bf, 2048, 2048, gwihr, 3072, nullptr, nullptr, nullptr, gi, 1);
  wmma_gemm<0, 0><<<dim3(1, 24), 256, 0, stream>>>(r1bf,     2048, 1024, gwhhr, 3072, nullptr, nullptr, nullptr, gh, 1);
  gru_pointwise<<<256, 256, 0, stream>>>(gi, gh, gbihr, gbhhr, r1f, 1024, r2f, r2bf, 2048);

  // --- final attention over rep[2,64,2048] (tanh activation) ---
  wmma_gemm<2, 1><<<dim3(2, 1), 256, 0, stream>>>(
      rep_bf, 2048, 2048, Wf_bf, 1024, bias_f, v_w, v_b, scores_f, 8);
  softmax_kernel<<<64, 256, 0, stream>>>(scores_f, out_attn, 2);
  context_kernel<0><<<dim3(64, 8), 256, 0, stream>>>(out_attn, rep_bf, out_ctx, 2, 2048);
}
